// FusedAttention_69183333204187
// MI455X (gfx1250) — matrix operand order repaired and checked by reading, not compiled, not run on hardware
//
#include <hip/hip_runtime.h>
#include <hip/hip_bf16.h>
#include <stdint.h>

// ---------------------------------------------------------------------------
// Fused multi-head causal attention for gfx1250 (MI455X), bf16 WMMA pipeline.
//   x[4,2048,1024] -> Q,K,V (x@W*) -> causal softmax(QK^T/sqrt(64))@V -> @Wp+bp
// ~137 GFLOP vs ~80MB traffic => compute bound; everything through
// v_wmma_f32_16x16x32_bf16 (f32 accumulate). Weight tiles for the QKV GEMM are
// staged into LDS by the Tensor Data Mover (tensor_load_to_lds, TENSORcnt),
// double-buffered and overlapped with compute.
// ---------------------------------------------------------------------------

typedef __attribute__((ext_vector_type(16))) __bf16 v16bf;
typedef __attribute__((ext_vector_type(8)))  __bf16 v8bf;
typedef __attribute__((ext_vector_type(8)))  float  v8f;
typedef __attribute__((ext_vector_type(4)))  unsigned int v4u;
typedef __attribute__((ext_vector_type(8)))  int    v8i;
typedef __attribute__((ext_vector_type(4)))  int    v4i;

#define N_EMBED   1024
#define HEAD_SIZE 64
#define NUM_HEADS 16
#define SEQ       2048
#define BATCH     4
#define MTOT      (BATCH * SEQ)   // 8192

// float -> bf16, round-to-nearest-even
__device__ __forceinline__ uint16_t f2bf(float f) {
  uint32_t u = __float_as_uint(f);
  uint32_t r = u + 0x7FFFu + ((u >> 16) & 1u);
  return (uint16_t)(r >> 16);
}

__device__ __forceinline__ v16bf cat16(v8bf lo, v8bf hi) {
  return __builtin_shufflevector(lo, hi, 0,1,2,3,4,5,6,7,8,9,10,11,12,13,14,15);
}

// A fragment (16x32, M x K), src row-major with stride lda, tile at (row0,k0).
// Layout used here (contiguous k, not ISA table order): lane m = lane&15, half = lane>>4;
//   k = half*16 + j (16 contiguous -> two b128)
__device__ __forceinline__ v16bf load_afrag(const uint16_t* src, int row0, int k0, int lda) {
  int lane = threadIdx.x & 31;
  int m = lane & 15, half = lane >> 4;
  const uint16_t* base = src + (size_t)(row0 + m) * lda + k0 + half * 16;
  v8bf lo = *(const v8bf*)(base);
  v8bf hi = *(const v8bf*)(base + 8);
  return cat16(lo, hi);
}

// B fragment (32x16, K x N) loaded from an N-major array (N x K row-major,
// stride ldb), tile at (n0, k0).
// Layout used here (contiguous k, not ISA table order): lane n = lane&15, half = lane>>4; k = half*16 + j (16 contiguous).
__device__ __forceinline__ v16bf load_bfrag(const uint16_t* src, int n0, int k0, int ldb) {
  int lane = threadIdx.x & 31;
  int n = lane & 15, half = lane >> 4;
  const uint16_t* base = src + (size_t)(n0 + n) * ldb + k0 + half * 16;
  v8bf lo = *(const v8bf*)(base);
  v8bf hi = *(const v8bf*)(base + 8);
  return cat16(lo, hi);
}

__device__ __forceinline__ v8f wmma_bf16(v16bf a, v16bf b, v8f c) {
  return __builtin_amdgcn_wmma_f32_16x16x32_bf16(false, a, false, b, (short)0, c, false, false);
}

// ---------------------------------------------------------------------------
// Tensor Data Mover: async-load a 2-D bf16 tile (tile_d0 contiguous elements x
// tile_d1 rows, row stride stride_d0 elements) from global into LDS at lds_off.
// D# per CDNA5 ISA 8.3/8.4: group0 = count|lds_addr|global_addr|type=2;
// group1 = data_size=2B, tensor dims == tile dims (no OOB), dim0 stride.
// Tracked on TENSORcnt.
// ---------------------------------------------------------------------------
__device__ __forceinline__ void tdm_load_2d(uint32_t lds_off, const void* gsrc,
                                            uint32_t tile_d0, uint32_t tile_d1,
                                            uint32_t stride_d0) {
  uint64_t ga = (uint64_t)(uintptr_t)gsrc;
  v4u g0 = { 1u,                                   // count=1, user descriptor
             lds_off,                              // lds_addr (bytes)
             (uint32_t)ga,                         // global_addr[31:0]
             (uint32_t)(ga >> 32) | (2u << 30) };  // global_addr[56:32] | type=2
  v8i g1 = { (int)(1u << 16),                      // data_size=1 -> 2 bytes/elem
             (int)((tile_d0 & 0xFFFFu) << 16),     // tensor_dim0[15:0]
             (int)(((tile_d0 >> 16) & 0xFFFFu) | ((tile_d1 & 0xFFFFu) << 16)),
             (int)(((tile_d1 >> 16) & 0xFFFFu) | (tile_d0 << 16)),  // tile_dim0
             (int)(tile_d1 & 0xFFFFu),             // tile_dim1 (tile_dim2=0)
             (int)stride_d0,                       // tensor_dim0_stride[31:0]
             0,                                    // stride hi / dim1_stride lo
             0 };
  v4i z4 = {};
#if defined(__clang_major__) && (__clang_major__ >= 23)
  v8i z8 = {};
  __builtin_amdgcn_tensor_load_to_lds(g0, g1, z4, z4, z8, 0);
#else
  __builtin_amdgcn_tensor_load_to_lds(g0, g1, z4, z4, 0);
#endif
}

// ---------------------------------------------------------------------------
// Prep kernels
// ---------------------------------------------------------------------------
__global__ void cvt_bf16_kernel(const float* __restrict__ in, uint16_t* __restrict__ out, int n) {
  int i = blockIdx.x * blockDim.x + threadIdx.x;
  if (i < n) out[i] = f2bf(in[i]);
}

// 1024x1024 f32 W[k][n] -> bf16 Wt[n][k]
__global__ void transpose_w_kernel(const float* __restrict__ W, uint16_t* __restrict__ Wt) {
  int idx = blockIdx.x * blockDim.x + threadIdx.x;  // over 1024*1024
  int n = idx >> 10, k = idx & 1023;
  Wt[idx] = f2bf(W[(size_t)k * N_EMBED + n]);
}

// ---------------------------------------------------------------------------
// QKV projection: x[8192,1024]bf16 @ W^T[1024,1024]bf16 (N-major).
// grid (64, 16, 3): z selects Q/K/V. Block = 256 thr = 8 waves; wave tile 16x64.
// The 64x32 weight tile shared by all 8 waves is TDM-loaded into LDS
// (double-buffered); waves read B-fragments from LDS, A-fragments from global.
// Q stored *pre-scaled by 1/sqrt(hs)* as [B,H,T,hs]; K as [B,H,T,hs];
// V stored transposed as [B,H,hs,T] so PV B-fragments are contiguous in key.
// ---------------------------------------------------------------------------
__global__ __launch_bounds__(256) void qkv_gemm_kernel(
    const uint16_t* __restrict__ xb,
    const uint16_t* __restrict__ wqt, const uint16_t* __restrict__ wkt,
    const uint16_t* __restrict__ wvt,
    uint16_t* __restrict__ qb, uint16_t* __restrict__ kb, uint16_t* __restrict__ vt) {
  __shared__ __align__(64) uint16_t bsh[2][64 * 32];   // double-buffered B tile
  int warp = threadIdx.x >> 5, lane = threadIdx.x & 31;
  int m0 = blockIdx.x * 128 + warp * 16;
  int n0 = blockIdx.y * 64;
  int mode = blockIdx.z;
  const uint16_t* wt = (mode == 0) ? wqt : (mode == 1 ? wkt : wvt);
  bool tdm_wave = (threadIdx.x < 32);                  // one TDM issue per WG

  if (tdm_wave) {
    tdm_load_2d((uint32_t)(uintptr_t)&bsh[0][0],
                wt + (size_t)n0 * N_EMBED, 32u, 64u, (uint32_t)N_EMBED);
    __builtin_amdgcn_s_wait_tensorcnt(0);
  }
  __syncthreads();

  v8f c0 = {}, c1 = {}, c2 = {}, c3 = {};
  for (int k0 = 0; k0 < N_EMBED; k0 += 32) {
    int cur = (k0 >> 5) & 1;
    bool more = (k0 + 32) < N_EMBED;
    if (tdm_wave && more) {                            // async fill next buffer
      tdm_load_2d((uint32_t)(uintptr_t)&bsh[cur ^ 1][0],
                  wt + (size_t)n0 * N_EMBED + k0 + 32, 32u, 64u, (uint32_t)N_EMBED);
    }
    if (more) {                                        // global_prefetch_b8 next A
      __builtin_prefetch(xb + (size_t)m0 * N_EMBED + k0 + 32, 0, 1);
    }
    v16bf a = load_afrag(xb, m0, k0, N_EMBED);
    const uint16_t* bt = &bsh[cur][0];                 // LDS tile, row stride 32
    c0 = wmma_bf16(a, load_bfrag(bt,  0, 0, 32), c0);
    c1 = wmma_bf16(a, load_bfrag(bt, 16, 0, 32), c1);
    c2 = wmma_bf16(a, load_bfrag(bt, 32, 0, 32), c2);
    c3 = wmma_bf16(a, load_bfrag(bt, 48, 0, 32), c3);
    if (tdm_wave && more) __builtin_amdgcn_s_wait_tensorcnt(0);
    __syncthreads();
  }

  int nlo = lane & 15, half = lane >> 4;
  float scale = (mode == 0) ? 0.125f : 1.0f;  // 1/sqrt(64) folded into Q
#pragma unroll
  for (int t = 0; t < 4; ++t) {
    v8f c = (t == 0) ? c0 : (t == 1) ? c1 : (t == 2) ? c2 : c3;
#pragma unroll
    for (int i = 0; i < 8; ++i) {
      int m = m0 + i + 8 * half;            // global row  (b*T + t)
      int n = n0 + t * 16 + nlo;            // global col  (h*64 + c)
      int b = m / SEQ, tt = m % SEQ;
      int h = n / HEAD_SIZE, ch = n % HEAD_SIZE;
      size_t bh = (size_t)b * NUM_HEADS + h;
      uint16_t val = f2bf(c[i] * scale);
      if (mode == 0)      qb[(bh * SEQ + tt) * HEAD_SIZE + ch] = val;
      else if (mode == 1) kb[(bh * SEQ + tt) * HEAD_SIZE + ch] = val;
      else                vt[(bh * HEAD_SIZE + ch) * SEQ + tt] = val;
    }
  }
}

// ---------------------------------------------------------------------------
// Flash-style causal attention. 1 wave = one 16-row q-tile; 32 keys per step.
// Block = 128 thr = 4 independent waves. P tile staged C-layout -> LDS ->
// A-layout for the PV WMMAs.
// ---------------------------------------------------------------------------
__global__ __launch_bounds__(128) void attn_kernel(
    const uint16_t* __restrict__ qb, const uint16_t* __restrict__ kb,
    const uint16_t* __restrict__ vt, uint16_t* __restrict__ attn) {
  __shared__ __align__(32) uint16_t pshare[4][16 * 32];
  int warp = threadIdx.x >> 5, lane = threadIdx.x & 31;
  int wgid = blockIdx.x * 4 + warp;          // 8192 waves total
  int qt = wgid & 127;                       // 128 q-tiles per (b,h)
  int bh = wgid >> 7;                        // 0..63
  int b = bh >> 4, h = bh & 15;
  int q0 = qt * 16;
  const uint16_t* Q = qb + (size_t)bh * SEQ * HEAD_SIZE;
  const uint16_t* K = kb + (size_t)bh * SEQ * HEAD_SIZE;
  const uint16_t* V = vt + (size_t)bh * HEAD_SIZE * SEQ;

  v16bf aq0 = load_afrag(Q, q0,  0, HEAD_SIZE);   // Q rows, hs 0..31
  v16bf aq1 = load_afrag(Q, q0, 32, HEAD_SIZE);   // Q rows, hs 32..63

  v8f o0 = {}, o1 = {}, o2 = {}, o3 = {};
  float mrow[8], lrow[8];
#pragma unroll
  for (int i = 0; i < 8; ++i) { mrow[i] = -1e30f; lrow[i] = 0.f; }
  int nlo = lane & 15, half = lane >> 4;
  uint16_t* pp = pshare[warp];
  int qend = q0 + 16;

  for (int kk = 0; kk < qend; kk += 32) {
    // scores for keys kk..kk+15 and kk+16..kk+31 (2 WMMAs each over hs=64)
    v8f s0 = {}, s1 = {};
    s0 = wmma_bf16(aq0, load_bfrag(K, kk,       0, HEAD_SIZE), s0);
    s0 = wmma_bf16(aq1, load_bfrag(K, kk,      32, HEAD_SIZE), s0);
    s1 = wmma_bf16(aq0, load_bfrag(K, kk + 16,  0, HEAD_SIZE), s1);
    s1 = wmma_bf16(aq1, load_bfrag(K, kk + 16, 32, HEAD_SIZE), s1);

#pragma unroll
    for (int i = 0; i < 8; ++i) {
      int q = q0 + i + 8 * half;             // this element's query row
      float x0 = (kk + nlo      > q) ? -1e30f : s0[i];   // causal mask
      float x1 = (kk + 16 + nlo > q) ? -1e30f : s1[i];
      float mx = fmaxf(x0, x1);
#pragma unroll
      for (int d = 8; d >= 1; d >>= 1) mx = fmaxf(mx, __shfl_xor(mx, d, 32));
      float mnew = fmaxf(mrow[i], mx);
      float corr = __expf(mrow[i] - mnew);
      float e0 = __expf(x0 - mnew);
      float e1 = __expf(x1 - mnew);
      float rs = e0 + e1;
#pragma unroll
      for (int d = 8; d >= 1; d >>= 1) rs += __shfl_xor(rs, d, 32);
      lrow[i] = lrow[i] * corr + rs;
      mrow[i] = mnew;
      o0[i] *= corr; o1[i] *= corr; o2[i] *= corr; o3[i] *= corr;
      // stage P (C layout: row i+8*half, cols nlo / nlo+16) into LDS tile
      pp[(i + 8 * half) * 32 + nlo]      = f2bf(e0);
      pp[(i + 8 * half) * 32 + 16 + nlo] = f2bf(e1);
    }
    // same-wave LDS RAW: compiler inserts s_wait_dscnt; re-read in A layout
    v16bf pa = load_afrag(pp, 0, 0, 32);
    o0 = wmma_bf16(pa, load_bfrag(V,  0, kk, SEQ), o0);
    o1 = wmma_bf16(pa, load_bfrag(V, 16, kk, SEQ), o1);
    o2 = wmma_bf16(pa, load_bfrag(V, 32, kk, SEQ), o2);
    o3 = wmma_bf16(pa, load_bfrag(V, 48, kk, SEQ), o3);
  }

#pragma unroll
  for (int i = 0; i < 8; ++i) {
    float inv = 1.0f / lrow[i];
    size_t m = (size_t)b * SEQ + q0 + i + 8 * half;
    size_t cb = (size_t)h * HEAD_SIZE;
    attn[m * N_EMBED + cb +  0 + nlo] = f2bf(o0[i] * inv);
    attn[m * N_EMBED + cb + 16 + nlo] = f2bf(o1[i] * inv);
    attn[m * N_EMBED + cb + 32 + nlo] = f2bf(o2[i] * inv);
    attn[m * N_EMBED + cb + 48 + nlo] = f2bf(o3[i] * inv);
  }
}

// ---------------------------------------------------------------------------
// Output projection: attn[8192,1024]bf16 @ Wp^T + bp -> f32 out
// ---------------------------------------------------------------------------
__global__ __launch_bounds__(256) void proj_gemm_kernel(
    const uint16_t* __restrict__ at, const uint16_t* __restrict__ wpt,
    const float* __restrict__ bp, float* __restrict__ out) {
  int warp = threadIdx.x >> 5, lane = threadIdx.x & 31;
  int m0 = blockIdx.x * 128 + warp * 16;
  int n0 = blockIdx.y * 64;
  v8f c0 = {}, c1 = {}, c2 = {}, c3 = {};
  for (int k0 = 0; k0 < N_EMBED; k0 += 32) {
    if (k0 + 32 < N_EMBED) {
      __builtin_prefetch(at + (size_t)m0 * N_EMBED + k0 + 32, 0, 1);
    }
    v16bf a = load_afrag(at, m0, k0, N_EMBED);
    c0 = wmma_bf16(a, load_bfrag(wpt, n0 +  0, k0, N_EMBED), c0);
    c1 = wmma_bf16(a, load_bfrag(wpt, n0 + 16, k0, N_EMBED), c1);
    c2 = wmma_bf16(a, load_bfrag(wpt, n0 + 32, k0, N_EMBED), c2);
    c3 = wmma_bf16(a, load_bfrag(wpt, n0 + 48, k0, N_EMBED), c3);
  }
  int nlo = lane & 15, half = lane >> 4;
#pragma unroll
  for (int i = 0; i < 8; ++i) {
    size_t m = (size_t)m0 + i + 8 * half;
    float* row = out + m * N_EMBED + n0;
    row[ 0 + nlo] = c0[i] + bp[n0 +  0 + nlo];
    row[16 + nlo] = c1[i] + bp[n0 + 16 + nlo];
    row[32 + nlo] = c2[i] + bp[n0 + 32 + nlo];
    row[48 + nlo] = c3[i] + bp[n0 + 48 + nlo];
  }
}

// ---------------------------------------------------------------------------
extern "C" void kernel_launch(void* const* d_in, const int* in_sizes, int n_in,
                              void* d_out, int out_size, void* d_ws, size_t ws_size,
                              hipStream_t stream) {
  const float* x  = (const float*)d_in[0];
  const float* Wq = (const float*)d_in[1];
  const float* Wk = (const float*)d_in[2];
  const float* Wv = (const float*)d_in[3];
  const float* Wp = (const float*)d_in[4];
  const float* bp = (const float*)d_in[5];
  float* out = (float*)d_out;

  char* ws = (char*)d_ws;
  const size_t MB = 1ull << 20;
  // xb (16MB) is reused as the attention-output buffer after QKV is done.
  uint16_t* xb  = (uint16_t*)(ws +  0 * MB);  // x bf16 [8192,1024]  / attn out
  uint16_t* wqt = (uint16_t*)(ws + 16 * MB);  // Wq^T bf16 [1024,1024]
  uint16_t* wkt = (uint16_t*)(ws + 18 * MB);
  uint16_t* wvt = (uint16_t*)(ws + 20 * MB);
  uint16_t* wpt = (uint16_t*)(ws + 22 * MB);
  uint16_t* qb  = (uint16_t*)(ws + 24 * MB);  // Q bf16 [B,H,T,hs] (pre-scaled)
  uint16_t* kb  = (uint16_t*)(ws + 40 * MB);  // K bf16 [B,H,T,hs]
  uint16_t* vt  = (uint16_t*)(ws + 56 * MB);  // V bf16 [B,H,hs,T]  (72MB total)

  // 1) precision prep
  cvt_bf16_kernel<<<(MTOT * N_EMBED) / 256, 256, 0, stream>>>(x, xb, MTOT * N_EMBED);
  transpose_w_kernel<<<(N_EMBED * N_EMBED) / 256, 256, 0, stream>>>(Wq, wqt);
  transpose_w_kernel<<<(N_EMBED * N_EMBED) / 256, 256, 0, stream>>>(Wk, wkt);
  transpose_w_kernel<<<(N_EMBED * N_EMBED) / 256, 256, 0, stream>>>(Wv, wvt);
  transpose_w_kernel<<<(N_EMBED * N_EMBED) / 256, 256, 0, stream>>>(Wp, wpt);

  // 2) QKV projections (z = 0/1/2 -> Q/K/V), TDM-staged weight tiles
  qkv_gemm_kernel<<<dim3(64, 16, 3), 256, 0, stream>>>(xb, wqt, wkt, wvt, qb, kb, vt);

  // 3) causal flash attention (8192 q-tiles, 4 waves/block)
  attn_kernel<<<2048, 128, 0, stream>>>(qb, kb, vt, xb);

  // 4) output projection + bias
  proj_gemm_kernel<<<dim3(64, 16), 256, 0, stream>>>(xb, wpt, bp, out);
}
